// _WaveRNNInferrer_21955872817232
// MI455X (gfx1250) — compile-verified, matrix operand order checked by reading, not run on hardware
//
#include <hip/hip_runtime.h>
#include <hip/hip_bf16.h>
#include <math.h>

typedef __attribute__((ext_vector_type(16))) _Float16 v16h;
typedef __attribute__((ext_vector_type(8)))  float    v8f;

#define HDIM   512
#define HALF   256
#define LFRM   60
#define FEAT   128
#define N3H    1536
#define BINS   256

// ---------------- workspace layout -----------------
// [0)                 : WT  f16 [96 ntiles][16 ksteps][32 lanes][16]  (=512x1536 repacked)
// [786432 halves)     : HEAD f16 [4][16 ntiles][8 ksteps][32][16]     (4 x 256x256 repacked)
// byte 2097152        : biasC f32 [60][768]
// byte 2281472        : biasF f32 [60][768]
// byte 2465792        : bc f32[768], bf f32[768]
#define HEAD_HALF_OFF  786432
#define BIASC_BYTE     2097152u
#define BIASF_BYTE     2281472u
#define BCBF_BYTE      2465792u

static __device__ inline v8f wmma_f16(v16h a, v16h b, v8f c) {
  return __builtin_amdgcn_wmma_f32_16x16x32_f16(false, a, false, b, (short)0, c,
                                                false, false);
}

// Build A operand holding vector x in matrix row 0 (lanes 0 and 16 carry the
// K-halves per the 16-bit A layout; all other lanes zero).
static __device__ inline v16h loadA_vec(const _Float16* x, int ks, int lane) {
  v16h a = {};
  if ((lane & 15) == 0) {
    int base = ks * 32 + ((lane < 16) ? 0 : 8);
#pragma unroll
    for (int i = 0; i < 8; ++i) {
      a[i]     = x[base + i];
      a[8 + i] = x[base + 16 + i];
    }
  }
  return a;
}

// 256x256 GEMV tile, B streamed from global (L2-resident). Split-K chains.
static __device__ inline float head_gemv(const v16h* __restrict__ Bp,
                                         const _Float16* __restrict__ x,
                                         int wave, int lane) {
  v8f acc0 = {}, acc1 = {};
#pragma unroll
  for (int ks = 0; ks < 8; ks += 2) {
    v16h a0 = loadA_vec(x, ks, lane);
    v16h a1 = loadA_vec(x, ks + 1, lane);
    v16h b0 = Bp[(wave * 8 + ks) * 32 + lane];
    v16h b1 = Bp[(wave * 8 + ks + 1) * 32 + lane];
    acc0 = wmma_f16(a0, b0, acc0);
    acc1 = wmma_f16(a1, b1, acc1);
  }
  return acc0[0] + acc1[0];
}

// 256x256 GEMV tile with register-resident B fragments (zero memory latency).
static __device__ inline float head_gemv_reg(const _Float16* __restrict__ x,
    int lane, v16h b0, v16h b1, v16h b2, v16h b3,
    v16h b4, v16h b5, v16h b6, v16h b7) {
  v8f acc0 = {}, acc1 = {};
  acc0 = wmma_f16(loadA_vec(x, 0, lane), b0, acc0);
  acc1 = wmma_f16(loadA_vec(x, 1, lane), b1, acc1);
  acc0 = wmma_f16(loadA_vec(x, 2, lane), b2, acc0);
  acc1 = wmma_f16(loadA_vec(x, 3, lane), b3, acc1);
  acc0 = wmma_f16(loadA_vec(x, 4, lane), b4, acc0);
  acc1 = wmma_f16(loadA_vec(x, 5, lane), b5, acc1);
  acc0 = wmma_f16(loadA_vec(x, 6, lane), b6, acc0);
  acc1 = wmma_f16(loadA_vec(x, 7, lane), b7, acc1);
  return acc0[0] + acc1[0];
}

// ============================================================================
// Kernel 1: repack weights into WMMA B-operand tile order (f16) + bias reorder
// ============================================================================
__global__ void __launch_bounds__(256) prep_kernel(
    const float* __restrict__ whh, const float* __restrict__ bhh,
    const float* __restrict__ cprew, const float* __restrict__ coutw,
    const float* __restrict__ fprew, const float* __restrict__ foutw,
    _Float16* __restrict__ wt, _Float16* __restrict__ headb,
    float* __restrict__ bcbf) {
  int idx = blockIdx.x * blockDim.x + threadIdx.x;
  if (idx < 786432) {
    int tile = idx >> 9, within = idx & 511;
    int lane = within >> 4, e = within & 15;
    int nt = tile >> 4, ks = tile & 15;
    int koff = (lane < 16) ? 0 : 8;
    int k = ks * 32 + koff + ((e < 8) ? e : e + 8);
    int n = nt * 16 + (lane & 15);
    int srcRow;
    if (n < 768) srcRow = (n >> 8) * 512 + (n & 255);
    else { int m = n - 768; srcRow = (m >> 8) * 512 + 256 + (m & 255); }
    wt[idx] = (_Float16)whh[srcRow * HDIM + k];
  } else if (idx < 786432 + 262144) {
    int j = idx - 786432;
    int m = j >> 16, r = j & 65535;
    int tile = r >> 9, within = r & 511;
    int lane = within >> 4, e = within & 15;
    int nt = tile >> 3, ks = tile & 7;
    int koff = (lane < 16) ? 0 : 8;
    int k = ks * 32 + koff + ((e < 8) ? e : e + 8);
    int n = nt * 16 + (lane & 15);
    const float* src = (m == 0) ? cprew : (m == 1) ? coutw : (m == 2) ? fprew : foutw;
    headb[j] = (_Float16)src[k * HALF + n];
  } else if (idx < 786432 + 262144 + 1536) {
    int j = idx - (786432 + 262144);
    if (j < 768) bcbf[j] = bhh[(j >> 8) * 512 + (j & 255)];
    else { int q = j - 768; bcbf[j] = bhh[(q >> 8) * 512 + 256 + (q & 255)]; }
  }
}

// ============================================================================
// Kernel 2: conditioning GEMM via WMMA, fold bias_ih, split coarse/fine.
// ============================================================================
__global__ void __launch_bounds__(32) cond_kernel(
    const float* __restrict__ lf, const float* __restrict__ cw,
    const float* __restrict__ cb, const float* __restrict__ bih,
    float* __restrict__ biasC, float* __restrict__ biasF) {
  int nt = blockIdx.x, mt = blockIdx.y;
  int lane = threadIdx.x;
  int lr = lane & 15;
  int koff = (lane < 16) ? 0 : 8;
  int colB = nt * 16 + lr;
  int rowA = mt * 16 + lr;
  v8f acc = {};
  for (int ks = 0; ks < 4; ++ks) {
    int K0 = ks * 32 + koff;
    v16h a = {}, b = {};
    if (rowA < LFRM) {
#pragma unroll
      for (int i = 0; i < 8; ++i) {
        a[i]     = (_Float16)lf[rowA * FEAT + K0 + i];
        a[8 + i] = (_Float16)lf[rowA * FEAT + K0 + 16 + i];
      }
    }
#pragma unroll
    for (int i = 0; i < 8; ++i) {
      b[i]     = (_Float16)cw[(K0 + i) * N3H + colB];
      b[8 + i] = (_Float16)cw[(K0 + 16 + i) * N3H + colB];
    }
    acc = wmma_f16(a, b, acc);
  }
#pragma unroll
  for (int vg = 0; vg < 8; ++vg) {
    int row = mt * 16 + vg + ((lane < 16) ? 0 : 8);
    int col = nt * 16 + lr;
    if (row < LFRM) {
      float v = acc[vg] + cb[col] + bih[col];
      int g = col >> 9, rem = col & 511;
      if (rem < 256) biasC[row * 768 + (g << 8) + rem] = v;
      else           biasF[row * 768 + (g << 8) + (rem - 256)] = v;
    }
  }
}

// ============================================================================
// Kernel 3: persistent sequential scan. One workgroup, 16 wave32.
// ============================================================================
__global__ void __launch_bounds__(512) scan_kernel(
    const _Float16* __restrict__ wt, const _Float16* __restrict__ headb,
    const float* __restrict__ biasC, const float* __restrict__ biasF,
    const float* __restrict__ bcbf,
    const float* __restrict__ wci, const float* __restrict__ wfi,
    const float* __restrict__ cpreb, const float* __restrict__ coutb,
    const float* __restrict__ fpreb, const float* __restrict__ foutb,
    const int* __restrict__ upsf, float* __restrict__ out) {
  __shared__ _Float16 sh_h16[HDIM];
  __shared__ float    sh_h[HDIM];
  __shared__ float    sh_hid[N3H];
  __shared__ _Float16 sh_va[HALF];
  __shared__ _Float16 sh_vb[HALF];
  __shared__ float    sh_pv[16];
  __shared__ int      sh_pi[16];
  __shared__ int      sh_state[4];    // c_last, f_last, c_new, f_new

  const int tid  = threadIdx.x;
  const int lane = tid & 31;
  const int wave = tid >> 5;
  const int U = upsf[0];
  const int T = LFRM * U;
  const v16h* wtv = (const v16h*)wt;
  const v16h* hb1 = (const v16h*)(headb + 1 * 65536);
  const v16h* hb3 = (const v16h*)(headb + 3 * 65536);

  // ---- hoist all step-invariant operands into registers ----
  const int o = tid & 255;
  const float wc_r0 = wci[o],        wc_r1 = wci[768 + o];
  const float wc_z0 = wci[256 + o],  wc_z1 = wci[1024 + o];
  const float wc_n0 = wci[512 + o],  wc_n1 = wci[1280 + o];
  const float wf_r0 = wfi[o],        wf_r1 = wfi[768 + o],  wf_r2 = wfi[1536 + o];
  const float wf_z0 = wfi[256 + o],  wf_z1 = wfi[1024 + o], wf_z2 = wfi[1792 + o];
  const float wf_n0 = wfi[512 + o],  wf_n1 = wfi[1280 + o], wf_n2 = wfi[2048 + o];
  const float bc_r = bcbf[o],        bc_z = bcbf[256 + o],  bc_n = bcbf[512 + o];
  const float bf_r = bcbf[768 + o],  bf_z = bcbf[1024 + o], bf_n = bcbf[1280 + o];
  const int   hn16 = wave * 16 + (lane & 15);
  const float pb_c = cpreb[hn16], ob_c = coutb[hn16];
  const float pb_f = fpreb[hn16], ob_f = foutb[hn16];

  // ---- preload the two "pre" head matrices' B fragments (register resident)
  const v16h* p0 = (const v16h*)(headb + 0 * 65536) + wave * 8 * 32 + lane;
  const v16h* p2 = (const v16h*)(headb + 2 * 65536) + wave * 8 * 32 + lane;
  const v16h c0 = p0[0],   c1 = p0[32],  c2 = p0[64],  c3 = p0[96];
  const v16h c4 = p0[128], c5 = p0[160], c6 = p0[192], c7 = p0[224];
  const v16h f0 = p2[0],   f1 = p2[32],  f2 = p2[64],  f3 = p2[96];
  const v16h f4 = p2[128], f5 = p2[160], f6 = p2[192], f7 = p2[224];

  if (tid < HDIM) { sh_h[tid] = 0.f; sh_h16[tid] = (_Float16)0.f; }
  if (tid == 0) { sh_state[0] = 128; sh_state[1] = 0; }
  __syncthreads();

  int frame = 0, tcnt = 0;
  float fC_r = 0.f, fC_z = 0.f, fC_n = 0.f, fF_r = 0.f, fF_z = 0.f, fF_n = 0.f;
  for (int t = 0; t < T; ++t) {
    // per-frame gate biases: reload only when the frame changes (every U steps)
    if (tcnt == 0) {
      const float* bC = biasC + frame * 768;
      const float* bF = biasF + frame * 768;
      fC_r = bC[o]; fC_z = bC[256 + o]; fC_n = bC[512 + o];
      fF_r = bF[o]; fF_z = bF[256 + o]; fF_n = bF[512 + o];
    }

    // ---- stage 1: hid[1536] = h @ [Wc^T | Wf^T]
    // wave w owns n-tiles w*6..w*6+5; one shared A fragment per k-step,
    // 6 independent WMMA accumulator chains in flight.
    {
      v8f acc0 = {}, acc1 = {}, acc2 = {}, acc3 = {}, acc4 = {}, acc5 = {};
      const v16h* base = wtv + (wave * 6 * 16) * 32 + lane;
      for (int ks = 0; ks < 16; ++ks) {
        v16h a = loadA_vec(sh_h16, ks, lane);
        v16h b0 = base[(0 * 16 + ks) * 32];
        v16h b1 = base[(1 * 16 + ks) * 32];
        v16h b2 = base[(2 * 16 + ks) * 32];
        v16h b3 = base[(3 * 16 + ks) * 32];
        v16h b4 = base[(4 * 16 + ks) * 32];
        v16h b5 = base[(5 * 16 + ks) * 32];
        acc0 = wmma_f16(a, b0, acc0);
        acc1 = wmma_f16(a, b1, acc1);
        acc2 = wmma_f16(a, b2, acc2);
        acc3 = wmma_f16(a, b3, acc3);
        acc4 = wmma_f16(a, b4, acc4);
        acc5 = wmma_f16(a, b5, acc5);
      }
      if (lane < 16) {
        int n0 = wave * 96 + lane;
        sh_hid[n0 +  0] = acc0[0];
        sh_hid[n0 + 16] = acc1[0];
        sh_hid[n0 + 32] = acc2[0];
        sh_hid[n0 + 48] = acc3[0];
        sh_hid[n0 + 64] = acc4[0];
        sh_hid[n0 + 80] = acc5[0];
      }
    }
    __syncthreads();

    // ---- stage 2: coarse GRU half
    float sc = (float)sh_state[0] * (1.f / 127.5f) - 1.f;
    float sf = (float)sh_state[1] * (1.f / 127.5f) - 1.f;
    if (tid < HALF) {
      float gr = fC_r + sc * wc_r0 + sf * wc_r1 + sh_hid[o]       + bc_r;
      float gz = fC_z + sc * wc_z0 + sf * wc_z1 + sh_hid[256 + o] + bc_z;
      float gn = fC_n + sc * wc_n0 + sf * wc_n1;
      float r = 1.f / (1.f + __expf(-gr));
      float z = 1.f / (1.f + __expf(-gz));
      float n = tanhf(gn + r * (sh_hid[512 + o] + bc_n));
      float hn = (1.f - z) * n + z * sh_h[o];
      sh_h[o] = hn; sh_h16[o] = (_Float16)hn; sh_va[o] = (_Float16)hn;
    }
    __syncthreads();

    // ---- stage 3: relu(hc @ pre_c + b)   (register-resident B)
    {
      float v = head_gemv_reg(sh_va, lane, c0, c1, c2, c3, c4, c5, c6, c7);
      if (lane < 16) {
        v += pb_c;
        sh_vb[hn16] = (_Float16)(v > 0.f ? v : 0.f);
      }
    }
    __syncthreads();

    // ---- stage 4: logits_c fused with per-wave argmax (lanes hold n order)
    {
      float v = head_gemv(hb1, sh_vb, wave, lane);
      float lv = -3.402823466e38f; int bi = 0;
      if (lane < 16) { lv = v + ob_c; bi = hn16; }
      for (int off = 8; off > 0; off >>= 1) {
        float ov = __shfl_down(lv, off, 16);
        int   oi = __shfl_down(bi, off, 16);
        if (ov > lv || (ov == lv && oi < bi)) { lv = ov; bi = oi; }
      }
      if (lane == 0) { sh_pv[wave] = lv; sh_pi[wave] = bi; }
    }
    __syncthreads();
    if (tid == 0) {
      float v = sh_pv[0]; int bi = sh_pi[0];
      for (int wv = 1; wv < 16; ++wv)
        if (sh_pv[wv] > v) { v = sh_pv[wv]; bi = sh_pi[wv]; }
      sh_state[2] = bi;
    }
    __syncthreads();

    // ---- stage 5: fine GRU half (conditioned on fresh coarse sample)
    float scn = (float)sh_state[2] * (1.f / 127.5f) - 1.f;
    if (tid < HALF) {
      float gr = fF_r + sc * wf_r0 + sf * wf_r1 + scn * wf_r2 + sh_hid[768 + o]  + bf_r;
      float gz = fF_z + sc * wf_z0 + sf * wf_z1 + scn * wf_z2 + sh_hid[1024 + o] + bf_z;
      float gn = fF_n + sc * wf_n0 + sf * wf_n1 + scn * wf_n2;
      float r = 1.f / (1.f + __expf(-gr));
      float z = 1.f / (1.f + __expf(-gz));
      float n = tanhf(gn + r * (sh_hid[1280 + o] + bf_n));
      float hn = (1.f - z) * n + z * sh_h[256 + o];
      sh_h[256 + o] = hn; sh_h16[256 + o] = (_Float16)hn; sh_va[o] = (_Float16)hn;
    }
    __syncthreads();

    // ---- stage 6: relu(hf @ pre_f + b)   (register-resident B)
    {
      float v = head_gemv_reg(sh_va, lane, f0, f1, f2, f3, f4, f5, f6, f7);
      if (lane < 16) {
        v += pb_f;
        sh_vb[hn16] = (_Float16)(v > 0.f ? v : 0.f);
      }
    }
    __syncthreads();

    // ---- stage 7: logits_f fused with per-wave argmax
    {
      float v = head_gemv(hb3, sh_vb, wave, lane);
      float lv = -3.402823466e38f; int bi = 0;
      if (lane < 16) { lv = v + ob_f; bi = hn16; }
      for (int off = 8; off > 0; off >>= 1) {
        float ov = __shfl_down(lv, off, 16);
        int   oi = __shfl_down(bi, off, 16);
        if (ov > lv || (ov == lv && oi < bi)) { lv = ov; bi = oi; }
      }
      if (lane == 0) { sh_pv[wave] = lv; sh_pi[wave] = bi; }
    }
    __syncthreads();
    if (tid == 0) {
      float v = sh_pv[0]; int bi = sh_pi[0];
      for (int wv = 1; wv < 16; ++wv)
        if (sh_pv[wv] > v) { v = sh_pv[wv]; bi = sh_pi[wv]; }
      sh_state[3] = bi;
      out[t]     = (float)sh_state[2];
      out[T + t] = (float)bi;
      sh_state[0] = sh_state[2];
      sh_state[1] = bi;
    }
    __syncthreads();

    if (++tcnt == U) { tcnt = 0; ++frame; }
  }

  if (tid < HDIM) out[2 * T + tid] = sh_h[tid];
}

// ============================================================================
extern "C" void kernel_launch(void* const* d_in, const int* in_sizes, int n_in,
                              void* d_out, int out_size, void* d_ws, size_t ws_size,
                              hipStream_t stream) {
  const float* lf    = (const float*)d_in[0];
  const float* cw    = (const float*)d_in[1];
  const float* cb    = (const float*)d_in[2];
  const float* whh   = (const float*)d_in[3];
  const float* bhh   = (const float*)d_in[4];
  const float* bih   = (const float*)d_in[5];
  const float* wci   = (const float*)d_in[6];
  const float* wfi   = (const float*)d_in[7];
  const float* cprew = (const float*)d_in[8];
  const float* cpreb = (const float*)d_in[9];
  const float* coutw = (const float*)d_in[10];
  const float* coutb = (const float*)d_in[11];
  const float* fprew = (const float*)d_in[12];
  const float* fpreb = (const float*)d_in[13];
  const float* foutw = (const float*)d_in[14];
  const float* foutb = (const float*)d_in[15];
  const int*   upsf  = (const int*)d_in[16];
  float* out = (float*)d_out;

  char* ws = (char*)d_ws;
  _Float16* wt    = (_Float16*)ws;
  _Float16* headb = (_Float16*)ws + HEAD_HALF_OFF;
  float* biasC = (float*)(ws + BIASC_BYTE);
  float* biasF = (float*)(ws + BIASF_BYTE);
  float* bcbf  = (float*)(ws + BCBF_BYTE);

  prep_kernel<<<4102, 256, 0, stream>>>(whh, bhh, cprew, coutw, fprew, foutw,
                                        wt, headb, bcbf);
  cond_kernel<<<dim3(96, 4), 32, 0, stream>>>(lf, cw, cb, bih, biasC, biasF);
  scan_kernel<<<1, 512, 0, stream>>>(wt, headb, biasC, biasF, bcbf,
                                     wci, wfi, cpreb, coutb, fpreb, foutb,
                                     upsf, out);
}
